// ClassicalSelfAttention_65481071400580
// MI455X (gfx1250) — compile-verified, matrix operand order checked by reading, not run on hardware
//
#include <hip/hip_runtime.h>
#include <math.h>

typedef __attribute__((ext_vector_type(16))) __bf16 v16bf;
typedef __attribute__((ext_vector_type(8)))  float  v8f;

#define DIM    2048
#define NOUT   6144   // 3*DIM
#define BATCH  32

static __device__ __forceinline__ float4 ld4(const float* p) {
    return *(const float4*)p;
}

// Pack two contiguous 8-float runs into the 16-element bf16 operand vector.
// Per CDNA5 16-bit A/B layout: element e holds K = e + 8*((e>>3) + lanegroup),
// i.e. e=0..7 -> run starting at k0+8g, e=8..15 -> run starting at k0+8g+16.
static __device__ __forceinline__ v16bf pack16(float4 a, float4 b, float4 c, float4 d) {
    v16bf v;
    v[0]  = (__bf16)a.x; v[1]  = (__bf16)a.y; v[2]  = (__bf16)a.z; v[3]  = (__bf16)a.w;
    v[4]  = (__bf16)b.x; v[5]  = (__bf16)b.y; v[6]  = (__bf16)b.z; v[7]  = (__bf16)b.w;
    v[8]  = (__bf16)c.x; v[9]  = (__bf16)c.y; v[10] = (__bf16)c.z; v[11] = (__bf16)c.w;
    v[12] = (__bf16)d.x; v[13] = (__bf16)d.y; v[14] = (__bf16)d.z; v[15] = (__bf16)d.w;
    return v;
}

// Stage 1: Q = x @ W_rot^T, K = x @ W_ent^T   (M=32, N=6144, K=2048)
// One wave computes a 32x16 output tile via two v_wmma_f32_16x16x32_bf16 per K-step.
// blockIdx.y selects which weight matrix / output buffer.
__global__ __launch_bounds__(256) void proj_wmma_kernel(
    const float* __restrict__ x,
    const float* __restrict__ W_rot,
    const float* __restrict__ W_ent,
    float* __restrict__ Q,
    float* __restrict__ Kp)
{
    const int lane = threadIdx.x & 31;
    const int wave = threadIdx.x >> 5;
    const int gw   = blockIdx.x * 8 + wave;       // 0..383 wave-tiles (N/16)
    const float* __restrict__ W   = blockIdx.y ? W_ent : W_rot;
    float* __restrict__       Out = blockIdx.y ? Kp    : Q;

    const int g  = lane >> 4;                     // lane group 0/1
    const int nl = lane & 15;
    const int n  = gw * 16 + nl;                  // output feature this lane owns (B side)

    const float* __restrict__ wrow = W + (size_t)n * DIM;          // B: row n of W (K contiguous)
    const float* __restrict__ xr0  = x + (size_t)nl * DIM;         // A tile0: m = nl
    const float* __restrict__ xr1  = x + (size_t)(nl + 16) * DIM;  // A tile1: m = nl+16

    v8f acc0 = {};
    v8f acc1 = {};

    for (int k0 = 0; k0 < DIM; k0 += 32) {
        const int ka = k0 + 8 * g;        // first 8-float run
        const int kb = ka + 16;           // second 8-float run

        v16bf Bv = pack16(ld4(wrow + ka), ld4(wrow + ka + 4),
                          ld4(wrow + kb), ld4(wrow + kb + 4));
        v16bf A0 = pack16(ld4(xr0 + ka),  ld4(xr0 + ka + 4),
                          ld4(xr0 + kb),  ld4(xr0 + kb + 4));
        v16bf A1 = pack16(ld4(xr1 + ka),  ld4(xr1 + ka + 4),
                          ld4(xr1 + kb),  ld4(xr1 + kb + 4));

        acc0 = __builtin_amdgcn_wmma_f32_16x16x32_bf16(false, A0, false, Bv,
                                                       (short)0, acc0, false, false);
        acc1 = __builtin_amdgcn_wmma_f32_16x16x32_bf16(false, A1, false, Bv,
                                                       (short)0, acc1, false, false);
    }

    // C/D layout: lane (nl, g) holds column n, rows m = r + 8*g (acc0) and m+16 (acc1)
#pragma unroll
    for (int r = 0; r < 8; ++r) {
        const int m = r + 8 * g;
        Out[(size_t)m        * NOUT + n] = acc0[r];
        Out[(size_t)(m + 16) * NOUT + n] = acc1[r];
    }
}

// Stage 2: fused scores -> online softmax -> weighted sum with x.
// One wave per (b, d) row; lanes stride over e = 0..2047. K[b] and x[b] are
// L2-resident (32 KB per batch), never materialize the (D,D) score matrix.
__global__ __launch_bounds__(256) void attn_softmax_kernel(
    const float* __restrict__ x,
    const float* __restrict__ Q,
    const float* __restrict__ Kp,
    float* __restrict__ out)
{
    const int lane = threadIdx.x & 31;
    const int row  = blockIdx.x * 8 + (threadIdx.x >> 5);   // 0 .. 32*2048-1
    const int b    = row >> 11;
    const int d    = row & 2047;

    const float* __restrict__ qp = Q  + (size_t)b * NOUT + 3 * d;
    const float  q0 = qp[0], q1 = qp[1], q2 = qp[2];
    const float* __restrict__ kb = Kp + (size_t)b * NOUT;
    const float* __restrict__ xb = x  + (size_t)b * DIM;

    const float scale = 0.022097086912079608f;   // 1/sqrt(2048)

    float m = -__builtin_inff();
    float l = 0.0f;
    float acc = 0.0f;

    for (int e = lane; e < DIM; e += 32) {
        const float k0 = kb[3 * e + 0];
        const float k1 = kb[3 * e + 1];
        const float k2 = kb[3 * e + 2];
        const float s  = (q0 * k0 + q1 * k1 + q2 * k2) * scale;
        const float mn = fmaxf(m, s);
        const float c  = __expf(m - mn);      // 0 on first iteration (m = -inf)
        const float p  = __expf(s - mn);
        l   = l * c + p;
        acc = acc * c + p * xb[e];
        m   = mn;
    }

    // cross-lane combine of (m, l, acc) — wave32
#pragma unroll
    for (int off = 16; off > 0; off >>= 1) {
        const float m2 = __shfl_xor(m, off, 32);
        const float l2 = __shfl_xor(l, off, 32);
        const float a2 = __shfl_xor(acc, off, 32);
        const float mn = fmaxf(m, m2);
        const float c1 = __expf(m - mn);
        const float c2 = __expf(m2 - mn);
        l   = l * c1 + l2 * c2;
        acc = acc * c1 + a2 * c2;
        m   = mn;
    }

    if (lane == 0)
        out[(size_t)b * DIM + d] = acc / l;
}

extern "C" void kernel_launch(void* const* d_in, const int* in_sizes, int n_in,
                              void* d_out, int out_size, void* d_ws, size_t ws_size,
                              hipStream_t stream) {
    (void)in_sizes; (void)n_in; (void)out_size; (void)ws_size;
    const float* x     = (const float*)d_in[0];
    const float* W_rot = (const float*)d_in[1];
    const float* W_ent = (const float*)d_in[2];
    float* out = (float*)d_out;

    float* Q  = (float*)d_ws;                        // 32*6144 f32
    float* Kp = Q + (size_t)BATCH * NOUT;            // 32*6144 f32  (1.5 MB total)

    // Stage 1: 384 wave-tiles per matrix, 8 waves/block -> 48 blocks, y = matrix select
    dim3 g1(48, 2, 1);
    proj_wmma_kernel<<<g1, 256, 0, stream>>>(x, W_rot, W_ent, Q, Kp);

    // Stage 2: 32*2048 rows, 8 waves/block -> 8192 blocks
    attn_softmax_kernel<<<8192, 256, 0, stream>>>(x, Q, Kp, out);
}
